// GPT_14224931684939
// MI455X (gfx1250) — matrix operand order repaired and checked by reading, not compiled, not run on hardware
//
#include <hip/hip_runtime.h>
#include <hip/hip_bf16.h>

typedef __attribute__((ext_vector_type(16))) __bf16 v16bf;
typedef __attribute__((ext_vector_type(2)))  __bf16 v2bf;
typedef __attribute__((ext_vector_type(8)))  float  v8f;

// ---------------------------------------------------------------- embedding
__global__ __launch_bounds__(256) void embed_kernel(
    const int* __restrict__ x, const float* __restrict__ wte,
    const float* __restrict__ wpe, float* __restrict__ h, int T_) {
  int row = blockIdx.x;            // b*T + t
  int t = row % T_;
  int tok = x[row];
  const float* src = wte + (size_t)tok * 768;
  const float* pos = wpe + (size_t)t * 768;
  float* dst = h + (size_t)row * 768;
  for (int e = threadIdx.x; e < 768; e += 256)
    dst[e] = src[e] + pos[e];
}

// ---------------------------------------------------------------- layernorm
__global__ __launch_bounds__(256) void ln_kernel(
    const float* __restrict__ x, const float* __restrict__ w,
    const float* __restrict__ b, float* __restrict__ y) {
  int row = blockIdx.x;
  const float* xr = x + (size_t)row * 768;
  float* yr = y + (size_t)row * 768;
  int tid = threadIdx.x;
  float v0 = xr[tid], v1 = xr[tid + 256], v2 = xr[tid + 512];
  __shared__ float rs[256], rss[256];
  rs[tid]  = v0 + v1 + v2;
  rss[tid] = v0 * v0 + v1 * v1 + v2 * v2;
  __syncthreads();
  for (int off = 128; off > 0; off >>= 1) {
    if (tid < off) { rs[tid] += rs[tid + off]; rss[tid] += rss[tid + off]; }
    __syncthreads();
  }
  float mean = rs[0] * (1.0f / 768.0f);
  float var  = rss[0] * (1.0f / 768.0f) - mean * mean;
  float rstd = rsqrtf(var + 1e-5f);
  yr[tid]       = (v0 - mean) * rstd * w[tid]       + b[tid];
  yr[tid + 256] = (v1 - mean) * rstd * w[tid + 256] + b[tid + 256];
  yr[tid + 512] = (v2 - mean) * rstd * w[tid + 512] + b[tid + 512];
}

// ---------------------------------------------------------------- WMMA GEMM
// C[M,N] = act(A[M,K] @ B + bias) + resid ; B is [K,N] (transB=0) or [N,K] (transB=1)
// Block tile 128x128, 8 waves (4 along M x 2 along N), wave tile 32x64 (2x4 WMMAs).
// LDS tiles are stored pre-swizzled into the WMMA register image
// ([subtile][lane][element]) so fragment loads are contiguous 32B per lane.
#define TM 128
#define TN 128
#define TK 32

__global__ __launch_bounds__(256) void gemm_kernel(
    const float* __restrict__ A, const float* __restrict__ Bw,
    const float* __restrict__ bias, const float* __restrict__ resid,
    float* __restrict__ C, int M, int N, int K, int transB, int act) {
  __shared__ alignas(32) __bf16 Asw[8][32][16];   // 8 m-subtiles
  __shared__ alignas(32) __bf16 Bsw[8][32][16];   // 8 n-subtiles
  int tid = threadIdx.x;
  int lane = tid & 31, w = tid >> 5;
  int lm = lane & 15;
  int wm = (w & 3) * 32;          // wave row offset in block tile
  int wn = (w >> 2) * 64;         // wave col offset in block tile
  int bm = blockIdx.y * TM;
  int bn = blockIdx.x * TN;

  v8f acc[2][4] = {};

  for (int k0 = 0; k0 < K; k0 += TK) {
    // ---- A tile 128x32 (f32 -> bf16), swizzled write
    {
      int r = tid >> 1;                  // 0..127
      int cbase = (tid & 1) * 16;        // 0 | 16
      const float* src = A + (size_t)(bm + r) * K + k0 + cbase;
      if (k0 + TK < K) __builtin_prefetch(src + TK, 0, 1);
      int mt = r >> 4, lmr = r & 15;
#pragma unroll
      for (int j = 0; j < 8; j++) {
        int k = cbase + 2 * j;
        v2bf p; p[0] = (__bf16)src[2 * j]; p[1] = (__bf16)src[2 * j + 1];
        int ln = lmr + 16 * ((k >> 4) & 1);
        int e  = k & 15;
        *(v2bf*)&Asw[mt][ln][e] = p;     // e is even -> 4B aligned
      }
    }
    // ---- B tile 32x128, swizzled write
    if (transB) {
      // per-column loader: contiguous K in global AND contiguous LDS write
      int n = tid & 127;
      int kb2 = (tid >> 7) * 16;         // 0 | 16
      int col = bn + n;
      int nt = n >> 4;
      int ln = (n & 15) + 16 * (kb2 >> 4);
      const float* src = Bw + (size_t)col * K + k0 + kb2;
      if (col < N && k0 + TK < K) __builtin_prefetch(src + TK, 0, 1);
      __bf16 tmp[16];
#pragma unroll
      for (int j = 0; j < 16; j++)
        tmp[j] = (col < N) ? (__bf16)src[j] : (__bf16)0.0f;
#pragma unroll
      for (int j = 0; j < 8; j++) {
        v2bf p; p[0] = tmp[2 * j]; p[1] = tmp[2 * j + 1];
        *(v2bf*)&Bsw[nt][ln][2 * j] = p;
      }
    } else {
      // per-row loader: contiguous N in global, scattered u16 LDS writes
      int k = tid >> 3;                  // 0..31
      int cbase = (tid & 7) * 16;        // 0..112
      const float* src = Bw + (size_t)(k0 + k) * N + bn + cbase;
      if (k0 + TK < K) __builtin_prefetch(src + (size_t)TK * N, 0, 1);
#pragma unroll
      for (int j = 0; j < 16; j++) {
        int n = cbase + j;
        float v = (bn + n < N) ? src[j] : 0.0f;
        Bsw[n >> 4][(n & 15) + 16 * (k >> 4)][k & 15] = (__bf16)v;
      }
    }
    __syncthreads();

    v16bf af[2], bfg[4];
#pragma unroll
    for (int mi = 0; mi < 2; mi++)
      af[mi] = *(const v16bf*)&Asw[(wm >> 4) + mi][lane][0];
#pragma unroll
    for (int ni = 0; ni < 4; ni++)
      bfg[ni] = *(const v16bf*)&Bsw[(wn >> 4) + ni][lane][0];
#pragma unroll
    for (int mi = 0; mi < 2; mi++)
#pragma unroll
      for (int ni = 0; ni < 4; ni++)
        acc[mi][ni] = __builtin_amdgcn_wmma_f32_16x16x32_bf16(
            false, af[mi], false, bfg[ni], (short)0, acc[mi][ni], false, false);
    __syncthreads();
  }

  int half = lane >> 4;
#pragma unroll
  for (int mi = 0; mi < 2; mi++) {
#pragma unroll
    for (int ni = 0; ni < 4; ni++) {
      int col = bn + wn + ni * 16 + lm;
      if (col >= N) continue;
#pragma unroll
      for (int r = 0; r < 8; r++) {
        int row = bm + wm + mi * 16 + r + 8 * half;
        float v = acc[mi][ni][r];
        if (bias) v += bias[col];
        if (act == 1) v = 0.5f * v * (1.0f + erff(v * 0.70710678118654752f));
        if (resid) v += resid[(size_t)row * N + col];
        C[(size_t)row * N + col] = v;
      }
    }
  }
}

// ---------------------------------------------------------------- attention
// qkv: [B,T,3E], head-major inside E. o: [B,T,E] (= [B,T,H,D]).
// Block = 4 waves, 64 query rows (one (b,h)); each wave owns 16 rows.
// Ks/Vs/Ps live in LDS pre-swizzled to the WMMA register image.
__global__ __launch_bounds__(128) void attn_kernel(
    const float* __restrict__ qkv, float* __restrict__ o, int T_) {
  const int E_ = 768, D_ = 64;
  int b = blockIdx.z, hh = blockIdx.y;
  int qb0 = blockIdx.x * 64;
  int tid = threadIdx.x;
  int lane = tid & 31, w = tid >> 5;
  int lm = lane & 15, half = lane >> 4;
  int qb = qb0 + w * 16;

  __shared__ alignas(32) __bf16 Ksw[2][2][32][16];  // [key-tile][d-half][lane][e]
  __shared__ alignas(32) __bf16 Vsw[4][32][16];     // [d-tile][lane][e]
  __shared__ alignas(32) __bf16 Psw[4][32][16];     // [wave][lane][e]

  // Q fragments (A layout), d split 0..31 / 32..63
  v16bf qa[2];
#pragma unroll
  for (int f = 0; f < 2; f++) {
    const float* qp = qkv + ((size_t)(b * T_ + qb + lm) * 3) * E_ + hh * D_ + f * 32;
#pragma unroll
    for (int e = 0; e < 16; e++) {
      int kk = e + 16 * half;
      qa[f][e] = (__bf16)qp[kk];
    }
  }

  v8f oacc[4] = {};
  float mrun[8], lrun[8];
#pragma unroll
  for (int r = 0; r < 8; r++) { mrun[r] = -1e30f; lrun[r] = 0.0f; }

  int kend = qb0 + 64;                       // causal bound for this block
  for (int kb = 0; kb < kend; kb += 32) {
    // cooperative K/V tile load: thread -> one key row, 16 d values
    {
      int r = tid >> 2;                      // key 0..31
      int cbase = (tid & 3) * 16;            // d block
      const float* kp = qkv + ((size_t)(b * T_ + kb + r) * 3 + 1) * E_ + hh * D_ + cbase;
      const float* vp = qkv + ((size_t)(b * T_ + kb + r) * 3 + 2) * E_ + hh * D_ + cbase;
      int kt = r >> 4;
      int dh = cbase >> 5;
      int kh = (cbase >> 4) & 1;
      int lnk = (r & 15) + 16 * kh;          // K-frag lane, contiguous e write
#pragma unroll
      for (int j = 0; j < 8; j++) {
        v2bf p; p[0] = (__bf16)kp[2 * j]; p[1] = (__bf16)kp[2 * j + 1];
        *(v2bf*)&Ksw[kt][dh][lnk][2 * j] = p;
      }
      int nt = cbase >> 4;
      int ev = r & 15;
#pragma unroll
      for (int j = 0; j < 16; j++)           // V: scattered u16 writes
        Vsw[nt][j + 16 * kt][ev] = (__bf16)vp[j];
    }
    __syncthreads();

    // S = Q @ K^T : 2 key tiles x 2 d-halves
    v8f s[2] = {};
#pragma unroll
    for (int kt = 0; kt < 2; kt++) {
#pragma unroll
      for (int dh = 0; dh < 2; dh++) {
        v16bf bfr = *(const v16bf*)&Ksw[kt][dh][lane][0];
        s[kt] = __builtin_amdgcn_wmma_f32_16x16x32_bf16(
            false, qa[dh], false, bfr, (short)0, s[kt], false, false);
      }
    }

    // scale + causal mask + online softmax (row = r + 8*half, col = lane&15)
    float alpha[8];
#pragma unroll
    for (int r = 0; r < 8; r++) {
      int q = qb + r + 8 * half;
      float mx = -1e30f;
#pragma unroll
      for (int kt = 0; kt < 2; kt++) {
        int key = kb + kt * 16 + lm;
        float v = s[kt][r] * 0.125f;         // 1/sqrt(64)
        if (key > q) v = -1e30f;
        s[kt][r] = v;
        mx = fmaxf(mx, v);
      }
#pragma unroll
      for (int m = 1; m < 16; m <<= 1) mx = fmaxf(mx, __shfl_xor(mx, m, 32));
      float mnew = fmaxf(mrun[r], mx);
      alpha[r] = expf(mrun[r] - mnew);
      mrun[r] = mnew;
      float rsum = 0.0f;
#pragma unroll
      for (int kt = 0; kt < 2; kt++) {
        float p = expf(s[kt][r] - mnew);
        s[kt][r] = p;
        rsum += p;
      }
#pragma unroll
      for (int m = 1; m < 16; m <<= 1) rsum += __shfl_xor(rsum, m, 32);
      lrun[r] = lrun[r] * alpha[r] + rsum;
      // P -> LDS in A-frag image: ln = row + 16*kt, e = col
#pragma unroll
      for (int kt = 0; kt < 2; kt++)
        Psw[w][(r + 8 * half) + 16 * kt][lm] = (__bf16)s[kt][r];
    }
#pragma unroll
    for (int ni = 0; ni < 4; ni++)
#pragma unroll
      for (int r = 0; r < 8; r++) oacc[ni][r] *= alpha[r];

    __syncthreads();   // order Ps writes before A-layout reads (cross-lane)

    // O += P @ V
    v16bf pa = *(const v16bf*)&Psw[w][lane][0];
#pragma unroll
    for (int ni = 0; ni < 4; ni++) {
      v16bf vb = *(const v16bf*)&Vsw[ni][lane][0];
      oacc[ni] = __builtin_amdgcn_wmma_f32_16x16x32_bf16(
          false, pa, false, vb, (short)0, oacc[ni], false, false);
    }
    __syncthreads();   // protect Ksw/Vsw before next tile load
  }

#pragma unroll
  for (int ni = 0; ni < 4; ni++) {
#pragma unroll
    for (int r = 0; r < 8; r++) {
      int q = qb + r + 8 * half;
      o[(size_t)(b * T_ + q) * E_ + hh * D_ + ni * 16 + lm] = oacc[ni][r] / lrun[r];
    }
  }
}

// ---------------------------------------------------------------- launcher
extern "C" void kernel_launch(void* const* d_in, const int* in_sizes, int n_in,
                              void* d_out, int out_size, void* d_ws, size_t ws_size,
                              hipStream_t stream) {
  const int Lc = 12, Hc = 12, Ec = 768, Vc = 50257, Tc = 1024, Bc = 2;
  const int M = Bc * Tc;  // 2048

  const int*   x    = (const int*)  d_in[0];
  const float* wte  = (const float*)d_in[1];
  const float* wpe  = (const float*)d_in[2];
  const float* caw  = (const float*)d_in[3];
  const float* cab  = (const float*)d_in[4];
  const float* cpw  = (const float*)d_in[5];
  const float* cpb  = (const float*)d_in[6];
  const float* fcw  = (const float*)d_in[7];
  const float* fcb  = (const float*)d_in[8];
  const float* mpw  = (const float*)d_in[9];
  const float* mpb  = (const float*)d_in[10];
  const float* l1w  = (const float*)d_in[11];
  const float* l1b  = (const float*)d_in[12];
  const float* l2w  = (const float*)d_in[13];
  const float* l2b  = (const float*)d_in[14];
  const float* lfw  = (const float*)d_in[15];
  const float* lfb  = (const float*)d_in[16];

  float* h    = (float*)d_ws;                  // [M,E]
  float* n    = h    + (size_t)M * Ec;         // [M,E]
  float* qkvb = n    + (size_t)M * Ec;         // [M,3E]
  float* ob   = qkvb + (size_t)M * 3 * Ec;     // [M,E]
  float* mb   = ob   + (size_t)M * Ec;         // [M,4E]

  embed_kernel<<<M, 256, 0, stream>>>(x, wte, wpe, h, Tc);

  for (int l = 0; l < Lc; l++) {
    ln_kernel<<<M, 256, 0, stream>>>(h, l1w + (size_t)l * Ec, l1b + (size_t)l * Ec, n);

    gemm_kernel<<<dim3(3 * Ec / TN, M / TM), 256, 0, stream>>>(
        n, caw + (size_t)l * Ec * 3 * Ec, cab + (size_t)l * 3 * Ec,
        nullptr, qkvb, M, 3 * Ec, Ec, 0, 0);

    attn_kernel<<<dim3(Tc / 64, Hc, Bc), 128, 0, stream>>>(qkvb, ob, Tc);

    gemm_kernel<<<dim3(Ec / TN, M / TM), 256, 0, stream>>>(
        ob, cpw + (size_t)l * Ec * Ec, cpb + (size_t)l * Ec,
        h, h, M, Ec, Ec, 0, 0);

    ln_kernel<<<M, 256, 0, stream>>>(h, l2w + (size_t)l * Ec, l2b + (size_t)l * Ec, n);

    gemm_kernel<<<dim3(4 * Ec / TN, M / TM), 256, 0, stream>>>(
        n, fcw + (size_t)l * Ec * 4 * Ec, fcb + (size_t)l * 4 * Ec,
        nullptr, mb, M, 4 * Ec, Ec, 0, 1);

    gemm_kernel<<<dim3(Ec / TN, M / TM), 256, 0, stream>>>(
        mb, mpw + (size_t)l * 4 * Ec * Ec, mpb + (size_t)l * Ec,
        h, h, M, Ec, 4 * Ec, 0, 0);
  }

  ln_kernel<<<M, 256, 0, stream>>>(h, lfw, lfb, n);

  gemm_kernel<<<dim3((Vc + TN - 1) / TN, M / TM), 256, 0, stream>>>(
      n, wte, nullptr, nullptr, (float*)d_out, M, Vc, Ec, 1, 0);
}
